// Density_Block_2216203124960
// MI455X (gfx1250) — matrix operand order repaired and checked by reading, not compiled, not run on hardware
//
#include <hip/hip_runtime.h>
#include <hip/hip_bf16.h>

typedef __attribute__((ext_vector_type(16))) _Float16 v16h;
typedef __attribute__((ext_vector_type(8)))  _Float16 v8h;
typedef __attribute__((ext_vector_type(8)))  float    v8f;
typedef __attribute__((ext_vector_type(4)))  float    v4f;

#define IND      128
#define OUTD     256
#define NGRIDF   255.0f
#define WSTRIDE  136                      // 128 + 8 pad halves; 272B = 17*16B (b128-aligned, bank-spread)
#define WT_BYTES (OUTD * WSTRIDE * 2)     // 69632 B
#define SMEM_BYTES (WT_BYTES + OUTD * 4)  // + bias (f32)
#define ROWS_PER_BLOCK 256                // 8 waves * 32 rows

union V16 { v16h v; v8h h[2]; };

__global__ __launch_bounds__(256)
void density_block_kernel(const float* __restrict__ t,
                          const float* __restrict__ x,
                          const float* __restrict__ w,
                          const float* __restrict__ bias,
                          float* __restrict__ out)
{
    extern __shared__ __align__(16) char smem[];
    _Float16* wT    = (_Float16*)smem;            // [OUTD][WSTRIDE] f16, transposed weight
    float*    biasS = (float*)(smem + WT_BYTES);  // [OUTD]

    const int tid = threadIdx.x;

    // One-time per block: weight [128][256] f32 -> wT [256][136] f16 (transposed).
    for (int i = tid; i < IND * OUTD; i += 256) {
        int k  = i >> 8;
        int nn = i & 255;
        wT[nn * WSTRIDE + k] = (_Float16)w[i];
    }
    if (tid < OUTD) biasS[tid] = bias[tid];
    __syncthreads();

    const int wave   = tid >> 5;
    const int lane   = tid & 31;
    const int l15    = lane & 15;
    const int hi     = lane >> 4;
    const int hiBase = hi << 4;

    const int rbase = blockIdx.x * ROWS_PER_BLOCK + wave * 32;  // 32 rows per wave (2 groups of 16)

    // ---- A fragments: 2 groups x 16 rows x K=128, f32 -> f16, WMMA A layout ----
    V16 afrag[2][4];
    #pragma unroll
    for (int g = 0; g < 2; ++g) {
        const int arow = rbase + g * 16 + l15;
        const float* xr = x + (size_t)arow * IND;
        #pragma unroll
        for (int kt = 0; kt < 4; ++kt) {
            const int b0 = kt * 32 + hi * 8;
            v4f f0 = *(const v4f*)(xr + b0);
            v4f f1 = *(const v4f*)(xr + b0 + 4);
            v4f f2 = *(const v4f*)(xr + b0 + 16);
            v4f f3 = *(const v4f*)(xr + b0 + 20);
            #pragma unroll
            for (int e = 0; e < 4; ++e) {
                afrag[g][kt].v[e]      = (_Float16)f0[e];
                afrag[g][kt].v[4 + e]  = (_Float16)f1[e];
                afrag[g][kt].v[8 + e]  = (_Float16)f2[e];
                afrag[g][kt].v[12 + e] = (_Float16)f3[e];
            }
        }
    }

    // ---- per-row grid setup (row = g*16 + v + 8*hi, replicated across the half) ----
    int   Ui[2][8], myTile[2][8];
    float inter[2][8];
    #pragma unroll
    for (int g = 0; g < 2; ++g) {
        #pragma unroll
        for (int v = 0; v < 8; ++v) {
            float tv = t[rbase + g * 16 + v + 8 * hi];
            float u  = ceilf(tv * NGRIDF);
            inter[g][v] = 1.0f - (u - tv * NGRIDF);
            int ui = (int)u;
            int li = ui - 1; if (li < 0) li = 0;
            Ui[g][v] = ui;
            // the single tile this lane must capture for this row (L and U columns
            // are distinct lanes unless li==ui, in which case they coincide)
            myTile[g][v] = (l15 == (li & 15)) ? (li >> 4)
                         : ((l15 == (ui & 15)) ? (ui >> 4) : 99);
        }
    }

    float Cref[2][8], mR[2][8], sR[2][8], cap[2][8];
    #pragma unroll
    for (int g = 0; g < 2; ++g)
        #pragma unroll
        for (int v = 0; v < 8; ++v) { Cref[g][v] = 0.f; mR[g][v] = -1e30f; sR[g][v] = 0.f; cap[g][v] = 0.f; }

    // ---- 16 column tiles: WMMA (B fragment shared by both row groups), shuffle-free softmax ----
    for (int nt = 0; nt < 16; ++nt) {
        const int ncol = nt * 16 + l15;
        v8f accs[2];
        accs[0] = (v8f){}; accs[1] = (v8f){};
        #pragma unroll
        for (int kt = 0; kt < 4; ++kt) {
            V16 bf;
            const _Float16* bp = &wT[ncol * WSTRIDE + kt * 32 + hi * 8];
            bf.h[0] = *(const v8h*)(bp);
            bf.h[1] = *(const v8h*)(bp + 16);
            accs[0] = __builtin_amdgcn_wmma_f32_16x16x32_f16(false, afrag[0][kt].v, false, bf.v,
                                                             (short)0, accs[0], false, false);
            accs[1] = __builtin_amdgcn_wmma_f32_16x16x32_f16(false, afrag[1][kt].v, false, bf.v,
                                                             (short)0, accs[1], false, false);
        }
        const float bn = biasS[ncol];
        if (nt == 0) {
            #pragma unroll
            for (int g = 0; g < 2; ++g)
                #pragma unroll
                for (int v = 0; v < 8; ++v) {
                    float lv = accs[g][v] + bn;
                    Cref[g][v] = lv; mR[g][v] = lv; sR[g][v] = 1.0f;       // exp(lv-lv)=1
                    if (myTile[g][v] == 0) cap[g][v] = lv;
                }
        } else {
            #pragma unroll
            for (int g = 0; g < 2; ++g)
                #pragma unroll
                for (int v = 0; v < 8; ++v) {
                    float lv = accs[g][v] + bn;
                    mR[g][v] = fmaxf(mR[g][v], lv);
                    sR[g][v] += __expf(lv - Cref[g][v]);                    // per-lane, no shuffles
                    if (myTile[g][v] == nt) cap[g][v] = lv;
                }
        }
    }

    // ---- finalize: one butterfly merge + two gathers per row ----
    #pragma unroll
    for (int g = 0; g < 2; ++g) {
        #pragma unroll
        for (int v = 0; v < 8; ++v) {
            float Mx = mR[g][v];
            #pragma unroll
            for (int off = 1; off < 16; off <<= 1) Mx = fmaxf(Mx, __shfl_xor(Mx, off, 32));
            float sr = sR[g][v] * __expf(Cref[g][v] - Mx);                  // rebase lane sum
            #pragma unroll
            for (int off = 1; off < 16; off <<= 1) sr += __shfl_xor(sr, off, 32);

            int ui = Ui[g][v];
            int li = ui - 1; if (li < 0) li = 0;
            float gl = __shfl(cap[g][v], hiBase + (li & 15), 32);
            float gu = __shfl(cap[g][v], hiBase + (ui & 15), 32);
            float pL = __expf(gl - Mx);
            float pU = __expf(gu - Mx);
            float res = (pL + (pU - pL) * inter[g][v]) / sr;
            if (l15 == v) out[rbase + g * 16 + v + 8 * hi] = res;
        }
    }
}

extern "C" void kernel_launch(void* const* d_in, const int* in_sizes, int n_in,
                              void* d_out, int out_size, void* d_ws, size_t ws_size,
                              hipStream_t stream) {
    const float* t    = (const float*)d_in[0];   // [N]
    const float* x    = (const float*)d_in[1];   // [N,128]
    const float* w    = (const float*)d_in[2];   // [128,256]
    const float* bias = (const float*)d_in[3];   // [256]
    float* out = (float*)d_out;                  // [N]

    const int N = in_sizes[0];                   // 524288
    const int grid = N / ROWS_PER_BLOCK;         // 2048

    density_block_kernel<<<grid, 256, SMEM_BYTES, stream>>>(t, x, w, bias, out);
}
